// SwinTransformerBlock_34634616275471
// MI455X (gfx1250) — compile-verified, matrix operand order checked by reading, not run on hardware
//
#include <hip/hip_runtime.h>
#include <hip/hip_bf16.h>

// ---------------------------------------------------------------------------
// Swin Transformer block for MI455X (gfx1250), wave32 + WMMA f16->f32.
// Async GLOBAL->LDS double-buffered GEMMs (ASYNCcnt), dense precomputed
// attention bias, windowed WMMA attention.
// ---------------------------------------------------------------------------

typedef __attribute__((ext_vector_type(16))) _Float16 v16h;
typedef __attribute__((ext_vector_type(8)))  _Float16 v8h;
typedef __attribute__((ext_vector_type(8)))  float    v8f;

#define TOK   50176      // 16 * 3136 tokens (= 1024 windows * 49)
#define CDIM  512
#define NWIN  1024       // 16 batches * 64 windows
#define NHEAD 16

// ---- CDNA5 async memory->LDS copy (GLOBAL_LOAD_ASYNC_TO_LDS_B128, ASYNCcnt).
// LDS address = low 32 bits of the generic pointer (ISA: flat->LDS truncation).
__device__ __forceinline__ void async_copy_b128(void* lds_ptr, const void* gptr) {
  unsigned int lds_addr = (unsigned int)(unsigned long long)lds_ptr;
  asm volatile("global_load_async_to_lds_b128 %0, %1, off"
               :: "v"(lds_addr), "v"(gptr)
               : "memory");
}
__device__ __forceinline__ void wait_async0() {
  asm volatile("s_wait_asynccnt 0x0" ::: "memory");
}
__device__ __forceinline__ void wait_async2() {
  asm volatile("s_wait_asynccnt 0x2" ::: "memory");
}

// ---- WMMA fragment loader (ISA 7.12.2 16-bit A layout; B stored col-major
// in LDS so the identical per-lane pattern applies). Two 16B loads per lane.
__device__ __forceinline__ v16h ld_frag(const _Float16* base, int ld) {
  const int lane = threadIdx.x & 31;
  const int r  = lane & 15;          // row (A) / column (B) within 16-tile
  const int kb = (lane >> 4) << 3;   // lanes 16-31 hold K+8 group
  const _Float16* p = base + r * ld + kb;
  v8h lo = *(const v8h*)p;           // K = kb .. kb+7
  v8h hi = *(const v8h*)(p + 16);    // K = 16+kb .. 16+kb+7
  v16h out;
#pragma unroll
  for (int i = 0; i < 8; ++i) { out[i] = lo[i]; out[i + 8] = hi[i]; }
  return out;
}

__device__ __forceinline__ v8f wmma16(const v16h& a, const v16h& b, const v8f& c) {
  return __builtin_amdgcn_wmma_f32_16x16x32_f16(false, a, false, b, (short)0, c,
                                                false, false);
}

// ---------------------------------------------------------------------------
// Weight convert + transpose: W[k][n] f32 -> Wt[n][k] f16 (K contiguous).
// ---------------------------------------------------------------------------
__global__ void wconv_kernel(const float* __restrict__ W, _Float16* __restrict__ Wt,
                             int K, int Nn) {
  size_t idx = (size_t)blockIdx.x * 256 + threadIdx.x;
  if (idx < (size_t)K * Nn) {
    int n = (int)(idx / K);
    int k = (int)(idx % K);
    Wt[idx] = (_Float16)W[(size_t)k * Nn + n];
  }
}

// ---------------------------------------------------------------------------
// Expand relative-position bias to dense (NH, 64, 64) with -inf padding.
// One-shot: 65536 elements; lives in L2 for the attention pass.
// ---------------------------------------------------------------------------
__global__ void bias_expand_kernel(const float* __restrict__ bt,
                                   float* __restrict__ Bx) {
  int idx = blockIdx.x * 256 + threadIdx.x;    // h*4096 + n*64 + m
  int h = idx >> 12;
  int n = (idx >> 6) & 63;
  int m = idx & 63;
  float v = -1.0e30f;
  if (n < 49 && m < 49) {
    int rel = (n / 7 - m / 7 + 6) * 13 + (n % 7 - m % 7 + 6);
    v = bt[rel * NHEAD + h];
  }
  Bx[idx] = v;
}

// ---------------------------------------------------------------------------
// LayerNorm. MODE 0: read token order with roll(-3,-3)+window partition,
// write windowed row order.  MODE 1: identity row mapping.
// One wave per row: 32 lanes * 16 floats = 512 channels.
// ---------------------------------------------------------------------------
template <int MODE>
__global__ __launch_bounds__(256) void ln_kernel(const float* __restrict__ X,
                                                 const float* __restrict__ g,
                                                 const float* __restrict__ be,
                                                 _Float16* __restrict__ Out) {
  const int wave = threadIdx.x >> 5;
  const int lane = threadIdx.x & 31;
  const int r = blockIdx.x * 8 + wave;      // destination row
  size_t src;
  if (MODE == 0) {
    int win = r / 49, n = r % 49;
    int b = win >> 6, wi = win & 63;
    int i = (wi >> 3) * 7 + n / 7;
    int j = (wi & 7) * 7 + n % 7;
    int hh = (i + 3) % 56;                  // roll(-3): window reads x[(i+3)%56]
    int ww = (j + 3) % 56;
    src = (size_t)(b * 3136 + hh * 56 + ww);
  } else {
    src = (size_t)r;
  }
  const float* xp = X + src * CDIM + lane * 16;
  float v[16];
#pragma unroll
  for (int t = 0; t < 4; ++t) {
    float4 f = *(const float4*)(xp + t * 4);
    v[4 * t] = f.x; v[4 * t + 1] = f.y; v[4 * t + 2] = f.z; v[4 * t + 3] = f.w;
  }
  float s = 0.f;
#pragma unroll
  for (int t = 0; t < 16; ++t) s += v[t];
#pragma unroll
  for (int off = 16; off > 0; off >>= 1) s += __shfl_xor(s, off, 32);
  const float mean = s * (1.f / 512.f);
  float q = 0.f;
#pragma unroll
  for (int t = 0; t < 16; ++t) { float d = v[t] - mean; q += d * d; }
#pragma unroll
  for (int off = 16; off > 0; off >>= 1) q += __shfl_xor(q, off, 32);
  const float rstd = rsqrtf(q * (1.f / 512.f) + 1e-5f);
  v8h o0, o1;
#pragma unroll
  for (int t = 0; t < 8; ++t) {
    int c = lane * 16 + t;
    o0[t] = (_Float16)((v[t] - mean) * rstd * g[c] + be[c]);
  }
#pragma unroll
  for (int t = 0; t < 8; ++t) {
    int c = lane * 16 + 8 + t;
    o1[t] = (_Float16)((v[8 + t] - mean) * rstd * g[c] + be[c]);
  }
  _Float16* op = Out + (size_t)r * CDIM + lane * 16;
  *(v8h*)op = o0;
  *(v8h*)(op + 8) = o1;
}

// ---------------------------------------------------------------------------
// WMMA GEMM: C[M,N] = A[M,K] * Bt[N,K]^T (+bias, +epilogue).
// 64x64 output tile, BK=32, 128 threads = 4 waves; wave w owns rows w*16..+15.
// K-slabs double-buffered with GLOBAL_LOAD_ASYNC_TO_LDS_B128 (ASYNCcnt).
// EPI: 0 -> f16 out, 1 -> GELU(exact erf) f16 out,
//      2 -> proj: window-reverse + roll(+3) remap, +residual, f32 out,
//      3 -> +residual, f32 out.
// ---------------------------------------------------------------------------
template <int EPI>
__global__ __launch_bounds__(128) void gemm_wmma(const _Float16* __restrict__ A,
                                                 const _Float16* __restrict__ Bt,
                                                 const float* __restrict__ bias,
                                                 void* __restrict__ Out,
                                                 const float* __restrict__ Res,
                                                 int Nc, int K) {
  __shared__ __align__(16) _Float16 As[2][64 * 32];
  __shared__ __align__(16) _Float16 Bs[2][64 * 32];
  const int tid  = threadIdx.x;
  const int wave = tid >> 5;
  const int lane = tid & 31;
  const int row0 = blockIdx.x * 64;
  const int col0 = blockIdx.y * 64;

  v8f acc[4] = {};
  const int lt_r = tid >> 2;               // 0..31, each thread owns one 16B chunk
  const int lt_k = (tid & 3) * 8;
  const _Float16* gA = A + (size_t)(row0 + lt_r) * K + lt_k;
  const _Float16* gB = Bt + (size_t)(col0 + lt_r) * K + lt_k;
  _Float16* lA0 = &As[0][lt_r * 32 + lt_k];
  _Float16* lB0 = &Bs[0][lt_r * 32 + lt_k];
  _Float16* lA1 = &As[1][lt_r * 32 + lt_k];
  _Float16* lB1 = &Bs[1][lt_r * 32 + lt_k];

  const int ksteps = K >> 5;
  // prime buffer 0
  async_copy_b128(lA0, gA);
  async_copy_b128(lB0, gB);
  for (int s_ = 0; s_ < ksteps; ++s_) {
    const int cur = s_ & 1;
    if (s_ + 1 < ksteps) {                 // stream next slab while computing
      async_copy_b128(cur ? lA0 : lA1, gA + (s_ + 1) * 32);
      async_copy_b128(cur ? lB0 : lB1, gB + (s_ + 1) * 32);
      wait_async2();                       // current slab complete (2 in flight)
    } else {
      wait_async0();
    }
    __syncthreads();
    v16h a = ld_frag(&As[cur][wave * 16 * 32], 32);
#pragma unroll
    for (int nt = 0; nt < 4; ++nt) {
      v16h b = ld_frag(&Bs[cur][nt * 16 * 32], 32);
      acc[nt] = wmma16(a, b, acc[nt]);
    }
    __syncthreads();                       // all reads done before overwrite
  }

  // Epilogue: C/D layout -> row = i + 8*(lane>=16), col = lane&15 (ISA 7.12.2).
  const int rb = (lane >> 4) * 8;
  const int cc = lane & 15;
#pragma unroll
  for (int nt = 0; nt < 4; ++nt) {
    const int col = col0 + nt * 16 + cc;
    const float bv = bias[col];
#pragma unroll
    for (int i = 0; i < 8; ++i) {
      const int row = row0 + wave * 16 + rb + i;
      float v = acc[nt][i] + bv;
      if (EPI == 0) {
        ((_Float16*)Out)[(size_t)row * Nc + col] = (_Float16)v;
      } else if (EPI == 1) {
        float gl = 0.5f * v * (1.f + erff(v * 0.70710678118654752f));
        ((_Float16*)Out)[(size_t)row * Nc + col] = (_Float16)gl;
      } else if (EPI == 2) {
        int win = row / 49, n = row % 49;
        int b = win >> 6, wi = win & 63;
        int ii = (wi >> 3) * 7 + n / 7;
        int jj = (wi & 7) * 7 + n % 7;
        int hh = (ii + 3) % 56;            // roll(+3) on window-reverse
        int ww = (jj + 3) % 56;
        size_t o = (size_t)(b * 3136 + hh * 56 + ww) * CDIM + col;
        ((float*)Out)[o] = v + Res[o];
      } else {
        size_t o = (size_t)row * Nc + col;
        ((float*)Out)[o] = v + Res[o];
      }
    }
  }
}

// ---------------------------------------------------------------------------
// Windowed attention: one workgroup (4 waves) per (window, head).
// S = q k^T (4 WMMAs, K=32=dh) -> +dense bias, f32 softmax -> P V (4 WMMAs).
// q/k staged via async GLOBAL->LDS; v transposed manually for the B operand.
// ---------------------------------------------------------------------------
__global__ __launch_bounds__(128) void attn_kernel(const _Float16* __restrict__ QKV,
                                                   const float* __restrict__ Bx,
                                                   _Float16* __restrict__ O) {
  __shared__ __align__(16) _Float16 qs[64 * 32];   // q rows (padded to 64)
  __shared__ __align__(16) _Float16 ks[64 * 32];   // k rows (B for q*k^T)
  __shared__ __align__(16) _Float16 vt[32 * 64];   // v transposed [d][m]
  __shared__ float            sbuf[4][16 * 64];    // per-wave S (f32)
  __shared__ __align__(16) _Float16 pbuf[4][16 * 64]; // per-wave P (f16)

  const int win  = blockIdx.x >> 4;
  const int h    = blockIdx.x & 15;
  const int tid  = threadIdx.x;
  const int wave = tid >> 5;
  const int lane = tid & 31;
  const size_t base = (size_t)win * 49 * 1536 + h * 32;

  { // stage q/k/v for this (window, head); rows 49..63 zero-padded
    const int r  = tid >> 1;
    const int d0 = (tid & 1) * 16;
    if (r < 49) {
      const _Float16* qp = QKV + base + (size_t)r * 1536 + d0;
      const _Float16* kp = qp + 512;
      const _Float16* vp = qp + 1024;
      async_copy_b128(&qs[r * 32 + d0],     qp);
      async_copy_b128(&qs[r * 32 + d0 + 8], qp + 8);
      async_copy_b128(&ks[r * 32 + d0],     kp);
      async_copy_b128(&ks[r * 32 + d0 + 8], kp + 8);
#pragma unroll
      for (int j = 0; j < 16; ++j) vt[(d0 + j) * 64 + r] = vp[j];
    } else {
      v8h z = {};
      *(v8h*)&qs[r * 32 + d0] = z; *(v8h*)&qs[r * 32 + d0 + 8] = z;
      *(v8h*)&ks[r * 32 + d0] = z; *(v8h*)&ks[r * 32 + d0 + 8] = z;
#pragma unroll
      for (int j = 0; j < 16; ++j) vt[(d0 + j) * 64 + r] = (_Float16)0.f;
    }
  }
  wait_async0();
  __syncthreads();

  // ---- S = q k^T : wave owns query rows [wave*16, wave*16+16)
  v8f s[4] = {};
  {
    v16h a = ld_frag(&qs[wave * 16 * 32], 32);
#pragma unroll
    for (int nt = 0; nt < 4; ++nt) {
      v16h b = ld_frag(&ks[nt * 16 * 32], 32);
      s[nt] = wmma16(a, b, s[nt]);
    }
  }
  // scale + dense relative-position bias (branch-free), spill for softmax
  const float scale = 0.17677669529663687f;   // 1/sqrt(32)
  const int rb = (lane >> 4) * 8;
  const int cc = lane & 15;
  const float* bh = Bx + (size_t)h * 4096;
#pragma unroll
  for (int nt = 0; nt < 4; ++nt) {
#pragma unroll
    for (int i = 0; i < 8; ++i) {
      const int n = wave * 16 + rb + i;      // query index
      const int m = nt * 16 + cc;            // key index
      sbuf[wave][(rb + i) * 64 + m] = s[nt][i] * scale + bh[n * 64 + m];
    }
  }
  __syncthreads();

  // ---- softmax: lanes 0..15 each own one row of this wave's 16 rows
  if (lane < 16) {
    float* rowp = &sbuf[wave][lane * 64];
    float mx = -3.0e38f;
#pragma unroll 8
    for (int m = 0; m < 64; ++m) mx = fmaxf(mx, rowp[m]);
    float sum = 0.f;
#pragma unroll 8
    for (int m = 0; m < 64; ++m) { float e = __expf(rowp[m] - mx); rowp[m] = e; sum += e; }
    const float inv = 1.f / sum;
#pragma unroll 8
    for (int m = 0; m < 64; ++m)
      pbuf[wave][lane * 64 + m] = (_Float16)(rowp[m] * inv);
  }
  __syncthreads();

  // ---- O = P V : K = 64 keys (2 steps), N = 32 channels (2 tiles)
  v8f o[2] = {};
#pragma unroll
  for (int ks_ = 0; ks_ < 2; ++ks_) {
    v16h a = ld_frag(&pbuf[wave][ks_ * 32], 64);
#pragma unroll
    for (int nt = 0; nt < 2; ++nt) {
      v16h b = ld_frag(&vt[(nt * 16) * 64 + ks_ * 32], 64);
      o[nt] = wmma16(a, b, o[nt]);
    }
  }
#pragma unroll
  for (int nt = 0; nt < 2; ++nt) {
#pragma unroll
    for (int i = 0; i < 8; ++i) {
      const int n = wave * 16 + rb + i;
      if (n < 49)
        O[((size_t)win * 49 + n) * CDIM + h * 32 + nt * 16 + cc] = (_Float16)o[nt][i];
    }
  }
}

// ---------------------------------------------------------------------------
// Host-side orchestration
// ---------------------------------------------------------------------------
extern "C" void kernel_launch(void* const* d_in, const int* in_sizes, int n_in,
                              void* d_out, int out_size, void* d_ws, size_t ws_size,
                              hipStream_t stream) {
  const float* x       = (const float*)d_in[0];
  const float* w_qkv   = (const float*)d_in[3];
  const float* b_qkv   = (const float*)d_in[4];
  const float* w_proj  = (const float*)d_in[5];
  const float* b_proj  = (const float*)d_in[6];
  const float* btab    = (const float*)d_in[7];
  const float* g1      = (const float*)d_in[8];
  const float* be1     = (const float*)d_in[9];
  const float* g2      = (const float*)d_in[10];
  const float* be2     = (const float*)d_in[11];
  const float* w1      = (const float*)d_in[12];
  const float* b1      = (const float*)d_in[13];
  const float* w2      = (const float*)d_in[14];
  const float* b2      = (const float*)d_in[15];
  float* out = (float*)d_out;

  char* ws = (char*)d_ws;
  size_t off = 0;
  auto alloc = [&](size_t bytes) -> void* {
    off = (off + 255) & ~(size_t)255;
    void* p = ws + off;
    off += bytes;
    return p;
  };
  _Float16* wqkvT = (_Float16*)alloc((size_t)1536 * 512 * 2);
  _Float16* wprojT= (_Float16*)alloc((size_t)512 * 512 * 2);
  _Float16* w1T   = (_Float16*)alloc((size_t)2048 * 512 * 2);
  _Float16* w2T   = (_Float16*)alloc((size_t)512 * 2048 * 2);
  float*    Bx    = (float*)   alloc((size_t)NHEAD * 64 * 64 * 4);
  _Float16* Xw    = (_Float16*)alloc((size_t)TOK * 512 * 2);
  _Float16* QKV   = (_Float16*)alloc((size_t)TOK * 1536 * 2);
  _Float16* Ow    = (_Float16*)alloc((size_t)TOK * 512 * 2);
  float*    X1    = (float*)   alloc((size_t)TOK * 512 * 4);
  _Float16* XN2   = (_Float16*)alloc((size_t)TOK * 512 * 2);
  _Float16* H1    = (_Float16*)alloc((size_t)TOK * 2048 * 2);

  // 1) weights -> f16 transposed [N][K]; dense attention bias
  wconv_kernel<<<(512 * 1536 + 255) / 256, 256, 0, stream>>>(w_qkv, wqkvT, 512, 1536);
  wconv_kernel<<<(512 * 512 + 255) / 256, 256, 0, stream>>>(w_proj, wprojT, 512, 512);
  wconv_kernel<<<(512 * 2048 + 255) / 256, 256, 0, stream>>>(w1, w1T, 512, 2048);
  wconv_kernel<<<(2048 * 512 + 255) / 256, 256, 0, stream>>>(w2, w2T, 2048, 512);
  bias_expand_kernel<<<(NHEAD * 64 * 64) / 256, 256, 0, stream>>>(btab, Bx);

  // 2) LN1 + roll + window partition (f16, windowed rows)
  ln_kernel<0><<<TOK / 8, 256, 0, stream>>>(x, g1, be1, Xw);

  // 3) QKV = Xw @ WqkvT + b   (50176 x 1536, K=512)
  gemm_wmma<0><<<dim3(TOK / 64, 1536 / 64), 128, 0, stream>>>(
      Xw, wqkvT, b_qkv, QKV, nullptr, 1536, 512);

  // 4) windowed attention (1024 windows x 16 heads)
  attn_kernel<<<NWIN * NHEAD, 128, 0, stream>>>(QKV, Bx, Ow);

  // 5) proj + window reverse + roll back + shortcut -> X1 (f32, token order)
  gemm_wmma<2><<<dim3(TOK / 64, 512 / 64), 128, 0, stream>>>(
      Ow, wprojT, b_proj, X1, x, 512, 512);

  // 6) LN2 (token order)
  ln_kernel<1><<<TOK / 8, 256, 0, stream>>>(X1, g2, be2, XN2);

  // 7) FFN1 + exact GELU (50176 x 2048, K=512)
  gemm_wmma<1><<<dim3(TOK / 64, 2048 / 64), 128, 0, stream>>>(
      XN2, w1T, b1, H1, nullptr, 2048, 512);

  // 8) FFN2 + residual -> output (50176 x 512, K=2048)
  gemm_wmma<3><<<dim3(TOK / 64, 512 / 64), 128, 0, stream>>>(
      H1, w2T, b2, out, X1, 512, 2048);
}
